// TransformerEncoder_89799176225331
// MI455X (gfx1250) — compile-verified, matrix operand order checked by reading, not run on hardware
//
#include <hip/hip_runtime.h>
#include <hip/hip_bf16.h>

// MI455X / gfx1250 graph-transformer encoder layer.
// GEMMs: v_wmma_f32_16x16x32_bf16 (bf16 A/B, f32 accum).
// Weights staged to LDS via global_load_async_to_lds_b128 (ASYNCcnt path).
// Workspace requirement: 4 * N * C * 4 bytes + 384KB  (~205 MB).

#define NN 100000
#define CC 128
#define HH 8
#define HD 16
#define EE 600000
#define MT (NN / 16)   // 6250 exact M-tiles

typedef __bf16 v16bf __attribute__((ext_vector_type(16)));
typedef __bf16 v8bf  __attribute__((ext_vector_type(8)));
typedef float  v8f   __attribute__((ext_vector_type(8)));

// ---- tile loaders -------------------------------------------------------
// 16-bit A/B tile layout (ISA 7.12.2): lane r=lane&15 holds row (A: M, B: N),
// half = lane>>4 selects interleaved K blocks:
//   halves 0..7  <- K = kb0 .. kb0+7   (kb0 = kc*32 + half*8)
//   halves 8..15 <- K = kb0+16 .. kb0+23

__device__ __forceinline__ v16bf load_tile_f32(const float* __restrict__ row, int kb0) {
    float4 f0 = *(const float4*)(row + kb0);
    float4 f1 = *(const float4*)(row + kb0 + 4);
    float4 f2 = *(const float4*)(row + kb0 + 16);
    float4 f3 = *(const float4*)(row + kb0 + 20);
    v16bf a;
    a[0]  = (__bf16)f0.x; a[1]  = (__bf16)f0.y; a[2]  = (__bf16)f0.z; a[3]  = (__bf16)f0.w;
    a[4]  = (__bf16)f1.x; a[5]  = (__bf16)f1.y; a[6]  = (__bf16)f1.z; a[7]  = (__bf16)f1.w;
    a[8]  = (__bf16)f2.x; a[9]  = (__bf16)f2.y; a[10] = (__bf16)f2.z; a[11] = (__bf16)f2.w;
    a[12] = (__bf16)f3.x; a[13] = (__bf16)f3.y; a[14] = (__bf16)f3.z; a[15] = (__bf16)f3.w;
    return a;
}

__device__ __forceinline__ v16bf load_tile_bf(const __bf16* __restrict__ row, int kb0) {
    v8bf lo = *(const v8bf*)(row + kb0);
    v8bf hi = *(const v8bf*)(row + kb0 + 16);
    return __builtin_shufflevector(lo, hi, 0,1,2,3,4,5,6,7,8,9,10,11,12,13,14,15);
}

#define WMMA_BF16(A, B, C) \
    __builtin_amdgcn_wmma_f32_16x16x32_bf16(false, (A), false, (B), (short)0, (C), false, false)

// ---- async stage of one 32KB bf16 weight matrix into LDS ----------------
// 256 threads x 8 chunks x 16B = 32768B.  Generic->LDS: low 32 bits of a
// shared-aperture address are the raw LDS byte offset (ISA 10.2).
__device__ __forceinline__ void stage_weight_async(__bf16* lds, const __bf16* g, int tid) {
    #pragma unroll
    for (int i = 0; i < 8; i++) {
        int off = (tid + i * 256) * 16;
        unsigned lds_addr = (unsigned)(size_t)((const char*)lds + off);
        const char* ga = (const char*)g + off;
        asm volatile("global_load_async_to_lds_b128 %0, %1, off"
                     :: "v"(lds_addr), "v"(ga) : "memory");
    }
    asm volatile("s_wait_asynccnt 0x0" ::: "memory");
}

// ---- weight convert + transpose: WT[o][i] = (bf16)W[i][o] ---------------
__global__ void gt_cvtT_kernel(const float* __restrict__ W, __bf16* __restrict__ WT,
                               int in_dim, int out_dim) {
    int idx = blockIdx.x * blockDim.x + threadIdx.x;
    if (idx >= in_dim * out_dim) return;
    int i = idx / out_dim, o = idx % out_dim;
    WT[(size_t)o * in_dim + i] = (__bf16)W[idx];
}

// ---- LayerNorm: one wave per 128-wide row -------------------------------
__global__ __launch_bounds__(256) void gt_ln_kernel(const float* __restrict__ x,
                                                    const float* __restrict__ g,
                                                    const float* __restrict__ b,
                                                    float* __restrict__ out, int n) {
    int wid  = (blockIdx.x * blockDim.x + threadIdx.x) >> 5;
    int lane = threadIdx.x & 31;
    if (wid >= n) return;
    const float* rp = x + (size_t)wid * CC;
    float4 v = *(const float4*)(rp + lane * 4);
    float s = v.x + v.y + v.z + v.w;
    #pragma unroll
    for (int m = 1; m < 32; m <<= 1) s += __shfl_xor(s, m, 32);
    float mu = s * (1.0f / CC);
    float dx = v.x - mu, dy = v.y - mu, dz = v.z - mu, dw = v.w - mu;
    float ss = dx*dx + dy*dy + dz*dz + dw*dw;
    #pragma unroll
    for (int m = 1; m < 32; m <<= 1) ss += __shfl_xor(ss, m, 32);
    float inv = rsqrtf(ss * (1.0f / CC) + 1e-5f);
    float4 gg = *(const float4*)(g + lane * 4);
    float4 bb = *(const float4*)(b + lane * 4);
    float4 o;
    o.x = dx * inv * gg.x + bb.x; o.y = dy * inv * gg.y + bb.y;
    o.z = dz * inv * gg.z + bb.z; o.w = dw * inv * gg.w + bb.w;
    *(float4*)(out + (size_t)wid * CC + lane * 4) = o;
}

// ---- fused Q/K/V projection: one wave per 16-row M-tile -----------------
// Each weight matrix async-staged into LDS once per workgroup (8 waves share).
__global__ __launch_bounds__(256) void gt_qkv_kernel(const float* __restrict__ h,
        const __bf16* __restrict__ WqT, const __bf16* __restrict__ WkT, const __bf16* __restrict__ WvT,
        const float* __restrict__ bq, const float* __restrict__ bk, const float* __restrict__ bv,
        float* __restrict__ Q, float* __restrict__ K, float* __restrict__ V, int mtiles) {
    __shared__ __align__(16) __bf16 wlds[128 * 128];
    int tid  = threadIdx.x;
    int mt   = blockIdx.x * 8 + (tid >> 5);
    bool active = (mt < mtiles);
    int lane = tid & 31, r = lane & 15, half = lane >> 4;

    v16bf A[4];
    if (active) {
        const float* arow = h + (size_t)(mt * 16 + r) * CC;
        #pragma unroll
        for (int kc = 0; kc < 4; kc++) A[kc] = load_tile_f32(arow, kc * 32 + half * 8);
    }

    const __bf16* Ws[3] = {WqT, WkT, WvT};
    const float*  Bs[3] = {bq, bk, bv};
    float*        Os[3] = {Q, K, V};
    for (int w = 0; w < 3; w++) {
        stage_weight_async(wlds, Ws[w], tid);
        __syncthreads();
        if (active) {
            #pragma unroll
            for (int nt = 0; nt < 8; nt++) {
                v8f acc = {};
                const __bf16* brow = wlds + (nt * 16 + r) * CC;
                #pragma unroll
                for (int kc = 0; kc < 4; kc++)
                    acc = WMMA_BF16(A[kc], load_tile_bf(brow, kc * 32 + half * 8), acc);
                float bias = Bs[w][nt * 16 + r];
                float* op = Os[w] + (size_t)(mt * 16) * CC + nt * 16 + r;
                #pragma unroll
                for (int j = 0; j < 8; j++) op[(j + half * 8) * CC] = acc[j] + bias;
            }
        }
        __syncthreads();   // before next stage overwrites wlds
    }
}

// ---- per-edge attention: one wave per edge ------------------------------
__global__ __launch_bounds__(256) void gt_edge_kernel(const int* __restrict__ src,
        const int* __restrict__ dst, const float* __restrict__ Q, const float* __restrict__ K,
        const float* __restrict__ V, float* __restrict__ aggr, int e_count) {
    int e = (blockIdx.x * blockDim.x + threadIdx.x) >> 5;
    if (e >= e_count) return;
    int lane = threadIdx.x & 31;
    int s = src[e], d = dst[e];
    float4 q = *(const float4*)(Q + (size_t)d * CC + lane * 4);
    float4 k = *(const float4*)(K + (size_t)s * CC + lane * 4);
    float4 v = *(const float4*)(V + (size_t)s * CC + lane * 4);
    // per-head (16-wide) dot, 4 lanes per head
    float p = q.x * k.x + q.y * k.y + q.z * k.z + q.w * k.w;
    p += __shfl_xor(p, 1, 32);
    p += __shfl_xor(p, 2, 32);
    p *= 0.25f;                            // 1/sqrt(HD)
    // softmax over the 8 heads (values replicated x4; xor 4/8/16 hits each head once)
    float m = p;
    m = fmaxf(m, __shfl_xor(m, 4, 32));
    m = fmaxf(m, __shfl_xor(m, 8, 32));
    m = fmaxf(m, __shfl_xor(m, 16, 32));
    float ex = __expf(p - m);
    float sum = ex;
    sum += __shfl_xor(sum, 4, 32);
    sum += __shfl_xor(sum, 8, 32);
    sum += __shfl_xor(sum, 16, 32);
    float w = ex / sum;
    float* ap = aggr + (size_t)d * CC + lane * 4;
    atomicAdd(ap + 0, v.x * w);
    atomicAdd(ap + 1, v.y * w);
    atomicAdd(ap + 2, v.z * w);
    atomicAdd(ap + 3, v.w * w);
}

// ---- output projection + residual: x2 = aggr @ Wo + bo + x --------------
__global__ __launch_bounds__(256) void gt_outproj_kernel(const float* __restrict__ aggr,
        const __bf16* __restrict__ WoT, const float* __restrict__ bo,
        const float* __restrict__ xres, float* __restrict__ x2, int mtiles) {
    __shared__ __align__(16) __bf16 wlds[128 * 128];
    int tid = threadIdx.x;
    int mt  = blockIdx.x * 8 + (tid >> 5);
    bool active = (mt < mtiles);
    int lane = tid & 31, r = lane & 15, half = lane >> 4;

    stage_weight_async(wlds, WoT, tid);

    v16bf A[4];
    if (active) {
        const float* arow = aggr + (size_t)(mt * 16 + r) * CC;
        #pragma unroll
        for (int kc = 0; kc < 4; kc++) A[kc] = load_tile_f32(arow, kc * 32 + half * 8);
    }
    __syncthreads();
    if (!active) return;

    #pragma unroll
    for (int nt = 0; nt < 8; nt++) {
        v8f acc = {};
        const __bf16* brow = wlds + (nt * 16 + r) * CC;
        #pragma unroll
        for (int kc = 0; kc < 4; kc++)
            acc = WMMA_BF16(A[kc], load_tile_bf(brow, kc * 32 + half * 8), acc);
        float bias = bo[nt * 16 + r];
        #pragma unroll
        for (int j = 0; j < 8; j++) {
            size_t row = (size_t)(mt * 16 + j + half * 8);
            x2[row * CC + nt * 16 + r] = acc[j] + bias + xres[row * CC + nt * 16 + r];
        }
    }
}

// ---- fused FFN: relu(h2@W1+b1)@W2 + b2 + x2, 16x512 activ staged in LDS -
__global__ __launch_bounds__(128) void gt_ffn_kernel(const float* __restrict__ h2,
        const float* __restrict__ x2, const __bf16* __restrict__ W1T,
        const float* __restrict__ b1, const __bf16* __restrict__ W2T,
        const float* __restrict__ b2, float* __restrict__ out, int mtiles) {
    __shared__ __align__(16) __bf16 act[16 * 512];
    int mt = blockIdx.x;
    if (mt >= mtiles) return;
    int wave = threadIdx.x >> 5, lane = threadIdx.x & 31;
    int r = lane & 15, half = lane >> 4;
    // stage 1: each wave computes a 16x128 slice of the 16x512 activation
    const float* arow = h2 + (size_t)(mt * 16 + r) * CC;
    v16bf A[4];
    #pragma unroll
    for (int kc = 0; kc < 4; kc++) A[kc] = load_tile_f32(arow, kc * 32 + half * 8);
    #pragma unroll
    for (int nt = 0; nt < 8; nt++) {
        int n = wave * 128 + nt * 16 + r;
        v8f acc = {};
        const __bf16* brow = W1T + (size_t)n * CC;
        if (nt < 7) __builtin_prefetch(W1T + (size_t)(n + 16) * CC, 0, 0);
        #pragma unroll
        for (int kc = 0; kc < 4; kc++)
            acc = WMMA_BF16(A[kc], load_tile_bf(brow, kc * 32 + half * 8), acc);
        float bias = b1[n];
        #pragma unroll
        for (int j = 0; j < 8; j++) {
            float t = acc[j] + bias;
            act[(j + half * 8) * 512 + n] = (__bf16)fmaxf(t, 0.0f);
        }
    }
    __syncthreads();
    // stage 2: each wave computes 2 of the 8 output N-tiles, K=512 from LDS
    #pragma unroll
    for (int nt2 = 0; nt2 < 2; nt2++) {
        int nt = wave * 2 + nt2;
        v8f acc = {};
        const __bf16* brow  = W2T + (size_t)(nt * 16 + r) * 512;
        const __bf16* arow2 = &act[r * 512];
        __builtin_prefetch(brow, 0, 0);
        #pragma unroll
        for (int kc = 0; kc < 16; kc++) {
            int kb0 = kc * 32 + half * 8;
            acc = WMMA_BF16(load_tile_bf(arow2, kb0), load_tile_bf(brow, kb0), acc);
        }
        int n = nt * 16 + r;
        float bias = b2[n];
        #pragma unroll
        for (int j = 0; j < 8; j++) {
            size_t row = (size_t)(mt * 16 + j + half * 8);
            out[row * CC + n] = acc[j] + bias + x2[row * CC + n];
        }
    }
}

// ---- launcher -----------------------------------------------------------
extern "C" void kernel_launch(void* const* d_in, const int* in_sizes, int n_in,
                              void* d_out, int out_size, void* d_ws, size_t ws_size,
                              hipStream_t stream) {
    const float* x   = (const float*)d_in[0];
    const int*   ei  = (const int*)d_in[1];           // [2,E]; src then dst
    const float* Wq  = (const float*)d_in[2];  const float* bq  = (const float*)d_in[3];
    const float* Wk  = (const float*)d_in[4];  const float* bk  = (const float*)d_in[5];
    const float* Wv  = (const float*)d_in[6];  const float* bv  = (const float*)d_in[7];
    const float* Wo  = (const float*)d_in[8];  const float* bo  = (const float*)d_in[9];
    const float* W1  = (const float*)d_in[10]; const float* b1  = (const float*)d_in[11];
    const float* W2  = (const float*)d_in[12]; const float* b2  = (const float*)d_in[13];
    const float* g1  = (const float*)d_in[14]; const float* be1 = (const float*)d_in[15];
    const float* g2  = (const float*)d_in[16]; const float* be2 = (const float*)d_in[17];
    float* out = (float*)d_out;

    size_t nc = (size_t)NN * CC;
    float* B0 = (float*)d_ws;      // h1, then aggr
    float* B1 = B0 + nc;           // Q, then x2
    float* B2 = B1 + nc;           // K, then h2
    float* B3 = B2 + nc;           // V
    __bf16* wb  = (__bf16*)(B3 + nc);
    __bf16* WqT = wb;
    __bf16* WkT = WqT + 128 * 128;
    __bf16* WvT = WkT + 128 * 128;
    __bf16* WoT = WvT + 128 * 128;
    __bf16* W1T = WoT + 128 * 128;   // [512][128]
    __bf16* W2T = W1T + 128 * 512;   // [128][512]

    // bf16-transpose all weights (tiny)
    gt_cvtT_kernel<<<(128*128 + 255) / 256, 256, 0, stream>>>(Wq, WqT, 128, 128);
    gt_cvtT_kernel<<<(128*128 + 255) / 256, 256, 0, stream>>>(Wk, WkT, 128, 128);
    gt_cvtT_kernel<<<(128*128 + 255) / 256, 256, 0, stream>>>(Wv, WvT, 128, 128);
    gt_cvtT_kernel<<<(128*128 + 255) / 256, 256, 0, stream>>>(Wo, WoT, 128, 128);
    gt_cvtT_kernel<<<(128*512 + 255) / 256, 256, 0, stream>>>(W1, W1T, 128, 512);
    gt_cvtT_kernel<<<(512*128 + 255) / 256, 256, 0, stream>>>(W2, W2T, 512, 128);

    // LN1 -> h1 (B0)
    gt_ln_kernel<<<(NN + 7) / 8, 256, 0, stream>>>(x, g1, be1, B0, NN);
    // Q/K/V projections (weights via async LDS staging)
    gt_qkv_kernel<<<(MT + 7) / 8, 256, 0, stream>>>(B0, WqT, WkT, WvT, bq, bk, bv,
                                                    B1, B2, B3, MT);
    // zero aggr (reuses B0), then edge attention scatter-add
    hipMemsetAsync(B0, 0, nc * sizeof(float), stream);
    gt_edge_kernel<<<(EE + 7) / 8, 256, 0, stream>>>(ei, ei + EE, B1, B2, B3, B0, EE);
    // x2 = aggr @ Wo + bo + x   (into B1)
    gt_outproj_kernel<<<(MT + 7) / 8, 256, 0, stream>>>(B0, WoT, bo, x, B1, MT);
    // LN2 -> h2 (B2)
    gt_ln_kernel<<<(NN + 7) / 8, 256, 0, stream>>>(B1, g2, be2, B2, NN);
    // fused FFN + residual -> out
    gt_ffn_kernel<<<MT, 128, 0, stream>>>(B2, B1, W1T, b1, W2T, b2, out, MT);
}
